// RITS_29746943492659
// MI455X (gfx1250) — matrix operand order repaired and checked by reading, not compiled, not run on hardware
//
#include <hip/hip_runtime.h>

// RITS / BRITS recurrent imputation on gfx1250 (MI455X).
// Persistent per-block recurrence: h,c live in registers across all T=128 steps.
// All matmuls via v_wmma_f32_16x16x32_f16 (f16 in, f32 accumulate).
// Weights pre-swizzled into WMMA B-fragment order (32B contiguous per lane),
// streamed from L2 each step via an opaque *offset* (prevents LICM spills
// while keeping global-addrspace provenance -> global_load_b128, not flat).

#define B_  1024
#define T_  128
#define D_  64
#define H_  256
#define MB  16   // batch rows per workgroup

typedef __attribute__((ext_vector_type(16))) _Float16 v16h;
typedef __attribute__((ext_vector_type(8)))  float    v8f;

__device__ __forceinline__ v8f wmma16(v16h a, v16h b, v8f c) {
  // 8 args: (neg_a, A, neg_b, B, c_mod, C, reuse_a, reuse_b)
  return __builtin_amdgcn_wmma_f32_16x16x32_f16(false, a, false, b, (short)0, c,
                                                false, false);
}

// Defeat LICM on loop-invariant weight loads without destroying the pointer's
// global-addrspace provenance: only the 32-bit element offset goes opaque.
__device__ __forceinline__ int opqi(int x) {
  asm("" : "+v"(x));
  return x;
}

// A-matrix fragment (16x32 f16) gather from LDS row-major [16][ld].
// ISA layout: lanes 0-15 (g=0): v0..3 = K 0..7, v4..7 = K 16..23;
//             lanes 16-31 (g=1): v0..3 = K 8..15, v4..7 = K 24..31.
// Contiguous 8-half runs -> compiler emits 2x ds_load_b128.
__device__ __forceinline__ v16h loadA(const _Float16* buf, int ld, int lane, int kbase) {
  int m = lane & 15, g = lane >> 4;
  const _Float16* row = buf + m * ld + kbase + g * 8;
  v16h a;
#pragma unroll
  for (int v = 0; v < 8; ++v) {
    int k = (v < 4) ? (2 * v) : (16 + 2 * (v - 4));
    a[2 * v]     = row[k];
    a[2 * v + 1] = row[k + 1];
  }
  return a;
}

// B-matrix fragment (32x16 f16), pre-swizzled so each lane's 16 elements are
// contiguous (32 bytes -> 2x global_load_b128 from L1/L2).
__device__ __forceinline__ v16h loadB(const _Float16* Bsw, int ntiles, int kt, int nt, int lane) {
  int off = ((kt * ntiles + nt) * 32 + lane) * 16;
  return *(const v16h*)(Bsw + opqi(off));
}

__device__ __forceinline__ float sigm(float x) {
  return 1.0f / (1.0f + __expf(-x));
}

// f32 -> f16 weight conversion + swizzle into WMMA B-fragment order.
// dst linear index = (((kt*ntiles + nt)*32 + lane)*16 + e); k = kt*32 + (e < 8 ? g*8 + e : 16 + g*8 + (e-8)).
__global__ void rits_prep(const float* __restrict__ src, _Float16* __restrict__ dst,
                          int K, int N, int srcCols, int transpose) {
  int idx = blockIdx.x * blockDim.x + threadIdx.x;
  if (idx >= K * N) return;
  int e = idx & 15, lane = (idx >> 4) & 31, f = idx >> 9;
  int ntiles = N >> 4;
  int nt = f % ntiles, kt = f / ntiles;
  int n = nt * 16 + (lane & 15);
  int g = lane >> 4;
  int k = kt * 32 + ((e < 8) ? (g * 8 + e) : (16 + g * 8 + (e - 8)));
  float val = transpose ? src[n * srcCols + k] : src[k * srcCols + n];
  dst[idx] = (_Float16)val;
}

__global__ __launch_bounds__(256, 1) void rits_main(
    const float* __restrict__ values, const float* __restrict__ masks,
    const float* __restrict__ deltas,
    const float* __restrict__ td_h_b, const float* __restrict__ td_x_b,
    const float* __restrict__ hist_b, const float* __restrict__ fr_b,
    const float* __restrict__ wc_b,  const float* __restrict__ lstm_b,
    const float* __restrict__ out_W, const float* __restrict__ out_b,
    const _Float16* __restrict__ W,
    float* __restrict__ y_out, float* __restrict__ imp_out) {
  // Swizzled weight slabs (f16 element offsets)
  const _Float16* B1 = W;            // td_h_W.T   [64 x 256]
  const _Float16* B2 = W + 16384;    // td_x_W.T   [64 x 64]
  const _Float16* B3 = W + 20480;    // hist_W     [256 x 64]
  const _Float16* B4 = W + 36864;    // fr_W.T     [64 x 64]
  const _Float16* B5 = W + 40960;    // wc_W       [128 x 64]
  const _Float16* B6 = W + 49152;    // lstm_k     [128 x 1024]
  const _Float16* B7 = W + 180224;   // lstm_rk    [256 x 1024]

  __shared__ float    xs[MB][D_], ms[MB][D_], xh32[MB][D_], al32[MB][D_];
  __shared__ _Float16 d16[MB][D_], xc16[MB][D_];
  __shared__ _Float16 hA[MB][H_];        // decayed h (A operand)
  __shared__ _Float16 Awc[MB][2 * D_];   // [gamma_x | m]
  __shared__ _Float16 Alstm[MB][2 * D_]; // [c_c | m]
  __shared__ float    hfin[MB][H_];

  const int tid = threadIdx.x, lane = tid & 31, wave = tid >> 5;
  const int r0 = blockIdx.x * MB;
  const int n0 = wave * 32;            // hidden slice owned by this wave
  const int cn = lane & 15;            // column within a 16-tile
  const int rbase = 8 * (lane >> 4);   // accumulator row base

  v8f hreg[2] = {};                    // h for cols [n0, n0+32)
  v8f creg[2] = {};                    // c for cols [n0, n0+32)

  for (int t = 0; t < T_; ++t) {
    // ---- Phase 1: stage x, m, d for this step --------------------------
    for (int i = tid; i < MB * D_; i += 256) {
      int r = i >> 6, c = i & 63;
      size_t gi = ((size_t)(r0 + r) * T_ + t) * D_ + c;
      float xv = values[gi], mv = masks[gi], dv = deltas[gi];
      xs[r][c] = xv;
      ms[r][c] = mv;
      d16[r][c] = (_Float16)dv;
      _Float16 mh = (_Float16)mv;
      Awc[r][D_ + c] = mh;
      Alstm[r][D_ + c] = mh;
      if (t + 1 < T_) {
        __builtin_prefetch(values + gi + D_, 0, 3);
        __builtin_prefetch(masks + gi + D_, 0, 3);
        __builtin_prefetch(deltas + gi + D_, 0, 3);
      }
    }
    __syncthreads();

    // ---- Phase 2: all waves: gamma_h + decayed h; waves 4..7: gamma_x ---
    {
      v16h aD[2];
#pragma unroll
      for (int kt = 0; kt < 2; ++kt) aD[kt] = loadA(&d16[0][0], D_, lane, kt * 32);
#pragma unroll
      for (int s = 0; s < 2; ++s) {
        int nt = (n0 >> 4) + s;
        v8f acc = {};
#pragma unroll
        for (int kt = 0; kt < 2; ++kt)
          acc = wmma16(aD[kt], loadB(B1, H_ / 16, kt, nt, lane), acc);
        int col = nt * 16 + cn;
        float bias = td_h_b[col];
#pragma unroll
        for (int j = 0; j < 8; ++j) {
          float gh = __expf(-fmaxf(acc[j] + bias, 0.0f));
          hA[rbase + j][col] = (_Float16)(hreg[s][j] * gh);
        }
      }
      if (wave >= 4) {  // gamma_x = exp(-relu(d @ td_x_W.T + b))
        int nt = wave - 4;
        int col = nt * 16 + cn;
        v8f accg = {};
#pragma unroll
        for (int kt = 0; kt < 2; ++kt)
          accg = wmma16(aD[kt], loadB(B2, D_ / 16, kt, nt, lane), accg);
        float xb = td_x_b[col];
#pragma unroll
        for (int j = 0; j < 8; ++j)
          Awc[rbase + j][col] = (_Float16)__expf(-fmaxf(accg[j] + xb, 0.0f));
      }
    }
    __syncthreads();

    // ---- Phase 3: waves 0..3: x_h, x_c; waves 4..7: alpha ---------------
    if (wave < 4) {
      int nt = wave;
      int col = nt * 16 + cn;
      v8f acc = {};
#pragma unroll
      for (int kt = 0; kt < 8; ++kt)
        acc = wmma16(loadA(&hA[0][0], H_, lane, kt * 32),
                     loadB(B3, D_ / 16, kt, nt, lane), acc);
      float hb = hist_b[col];
#pragma unroll
      for (int j = 0; j < 8; ++j) {
        int row = rbase + j;
        float xh = acc[j] + hb;
        xh32[row][col] = xh;
        float m = ms[row][col], x = xs[row][col];
        xc16[row][col] = (_Float16)(m * x + (1.0f - m) * xh);
      }
    } else {  // alpha = [gamma_x | m] @ wc_W + b
      int nt = wave - 4;
      int col = nt * 16 + cn;
      v8f acca = {};
#pragma unroll
      for (int kt = 0; kt < 4; ++kt)
        acca = wmma16(loadA(&Awc[0][0], 2 * D_, lane, kt * 32),
                      loadB(B5, D_ / 16, kt, nt, lane), acca);
      float wcb = wc_b[col];
#pragma unroll
      for (int j = 0; j < 8; ++j)
        al32[rbase + j][col] = acca[j] + wcb;
    }
    __syncthreads();

    // ---- Phase 4 (waves 0..3): z_h, c_h, c_c (imputation out) -----------
    if (wave < 4) {
      int nt = wave, col = nt * 16 + cn;
      v8f accz = {};
#pragma unroll
      for (int kt = 0; kt < 2; ++kt)
        accz = wmma16(loadA(&xc16[0][0], D_, lane, kt * 32),
                      loadB(B4, D_ / 16, kt, nt, lane), accz);
      float frb = fr_b[col];
#pragma unroll
      for (int j = 0; j < 8; ++j) {
        int row = rbase + j;
        float zh = accz[j] + frb;
        float al = al32[row][col];
        float ch = al * zh + (1.0f - al) * xh32[row][col];
        float m = ms[row][col], x = xs[row][col];
        float cc = m * x + (1.0f - m) * ch;
        Alstm[row][col] = (_Float16)cc;
        imp_out[((size_t)(r0 + row) * T_ + t) * D_ + col] = cc;
      }
    }
    __syncthreads();

    // ---- Phase 5 (all waves): LSTM gates for own 32-wide hidden slice ---
    // Two sub-tile passes (s) to keep only 4 gate accumulators live;
    // hA fragments re-gathered from LDS (2x ds_load_b128 each) to cap VGPRs.
    {
      v16h aK[4];
#pragma unroll
      for (int kt = 0; kt < 4; ++kt) aK[kt] = loadA(&Alstm[0][0], 2 * D_, lane, kt * 32);
#pragma unroll
      for (int s = 0; s < 2; ++s) {
        v8f zg[4];
#pragma unroll
        for (int g = 0; g < 4; ++g) {
          int nt = (g * H_ + n0 + s * 16) >> 4;  // ntiles = 64
          v8f acc = {};
#pragma unroll
          for (int kt = 0; kt < 4; ++kt)
            acc = wmma16(aK[kt], loadB(B6, 64, kt, nt, lane), acc);
#pragma unroll
          for (int kt = 0; kt < 8; ++kt)
            acc = wmma16(loadA(&hA[0][0], H_, lane, kt * 32),
                         loadB(B7, 64, kt, nt, lane), acc);
          zg[g] = acc;
        }
        int hcol = n0 + s * 16 + cn;
        float bi = lstm_b[hcol], bf = lstm_b[H_ + hcol];
        float bg = lstm_b[2 * H_ + hcol], bo = lstm_b[3 * H_ + hcol];
#pragma unroll
        for (int j = 0; j < 8; ++j) {
          float iv = sigm(zg[0][j] + bi);
          float fv = sigm(zg[1][j] + bf);
          float gv = tanhf(zg[2][j] + bg);
          float ov = sigm(zg[3][j] + bo);
          float cw = fv * creg[s][j] + iv * gv;
          creg[s][j] = cw;
          hreg[s][j] = ov * tanhf(cw);
        }
      }
    }
    __syncthreads();
  }

  // ---- Epilogue: y = h @ out_W + out_b --------------------------------
#pragma unroll
  for (int s = 0; s < 2; ++s)
#pragma unroll
    for (int j = 0; j < 8; ++j)
      hfin[rbase + j][n0 + s * 16 + cn] = hreg[s][j];
  __syncthreads();
  if (tid < MB) {
    float acc = out_b[0];
    for (int k = 0; k < H_; ++k) acc += hfin[tid][k] * out_W[k];
    y_out[r0 + tid] = acc;
  }
}

extern "C" void kernel_launch(void* const* d_in, const int* in_sizes, int n_in,
                              void* d_out, int out_size, void* d_ws, size_t ws_size,
                              hipStream_t stream) {
  (void)in_sizes; (void)n_in; (void)out_size; (void)ws_size;
  const float* values  = (const float*)d_in[0];
  const float* masks   = (const float*)d_in[1];
  const float* deltas  = (const float*)d_in[2];
  const float* td_h_W  = (const float*)d_in[3];
  const float* td_h_b  = (const float*)d_in[4];
  const float* td_x_W  = (const float*)d_in[5];
  const float* td_x_b  = (const float*)d_in[6];
  const float* hist_W  = (const float*)d_in[7];
  const float* hist_b  = (const float*)d_in[8];
  const float* fr_W    = (const float*)d_in[9];
  const float* fr_b    = (const float*)d_in[10];
  const float* wc_W    = (const float*)d_in[11];
  const float* wc_b    = (const float*)d_in[12];
  const float* lstm_k  = (const float*)d_in[13];
  const float* lstm_rk = (const float*)d_in[14];
  const float* lstm_b  = (const float*)d_in[15];
  const float* out_W   = (const float*)d_in[16];
  const float* out_b   = (const float*)d_in[17];

  _Float16* W = (_Float16*)d_ws;
  _Float16* B1 = W;
  _Float16* B2 = W + 16384;
  _Float16* B3 = W + 20480;
  _Float16* B4 = W + 36864;
  _Float16* B5 = W + 40960;
  _Float16* B6 = W + 49152;
  _Float16* B7 = W + 180224;

  auto prep = [&](const float* src, _Float16* dst, int K, int N, int srcCols, int tr) {
    int total = K * N;
    rits_prep<<<(total + 255) / 256, 256, 0, stream>>>(src, dst, K, N, srcCols, tr);
  };
  prep(td_h_W,  B1,  64,  256,   64, 1);  // d @ td_h_W.T
  prep(td_x_W,  B2,  64,   64,   64, 1);  // d @ td_x_W.T
  prep(hist_W,  B3, 256,   64,   64, 0);  // h @ hist_W
  prep(fr_W,    B4,  64,   64,   64, 1);  // x_c @ fr_W.T
  prep(wc_W,    B5, 128,   64,   64, 0);  // [gx,m] @ wc_W
  prep(lstm_k,  B6, 128, 1024, 1024, 0);  // [c_c,m] @ lstm_k
  prep(lstm_rk, B7, 256, 1024, 1024, 0);  // h @ lstm_rk

  float* y_out = (float*)d_out;        // [B,1] first in tuple order
  float* imp_out = y_out + B_;         // then [B,T,D]
  rits_main<<<B_ / MB, 256, 0, stream>>>(values, masks, deltas, td_h_b, td_x_b,
                                         hist_b, fr_b, wc_b, lstm_b, out_W, out_b,
                                         W, y_out, imp_out);
}
